// MultiHeadAttentionLayer_87462714015771
// MI455X (gfx1250) — matrix operand order repaired and checked by reading, not compiled, not run on hardware
//
#include <hip/hip_runtime.h>
#include <stdint.h>

typedef __bf16 bf16;
typedef __attribute__((ext_vector_type(16))) __bf16 v16bf;
typedef __attribute__((ext_vector_type(8)))  __bf16 v8bf;
typedef __attribute__((ext_vector_type(8)))  float  v8f;

#define NUM_HEADS 8
#define DHEAD     64
#define IN_DIM    512
#define BATCH     4
#define SEQ       256
#define HD        512          /* NUM_HEADS*DHEAD */
#define QKV_LD    1536         /* Q|K|V row stride */

/* workspace layout (bytes) */
#define WS_WQKVT  0                            /* bf16 [1536][512]  (transposed Wq|Wk|Wv) */
#define WS_WET    (WS_WQKVT + 1536*512*2)      /* bf16 [512][512]   (transposed We)      */
#define WS_QKV    (WS_WET   + 512*512*2)       /* f32  [1024][1536]                      */
#define WS_S      (WS_QKV   + 1024*1536*4)     /* f32  [4][256][256][8] logits           */

/* ---- CDNA5 async global->LDS copy (ASYNCcnt-tracked, bypasses VGPRs) ---- */
__device__ __forceinline__ void async_b128(uint32_t ldsAddr, uint32_t vOff, uint64_t base)
{
    asm volatile("global_load_async_to_lds_b128 %0, %1, %2"
                 :: "v"(ldsAddr), "v"(vOff), "s"(base) : "memory");
}
/* wait until all but the newest stage (6 copies) have landed in LDS */
__device__ __forceinline__ void wait_async6()
{
    asm volatile("s_wait_asynccnt 0x6" ::: "memory");
}
__device__ __forceinline__ void wait_async0()
{
    asm volatile("s_wait_asynccnt 0x0" ::: "memory");
}

/* ---------------- weight conversion: fp32 -> bf16, transposed ---------------- */
__global__ void __launch_bounds__(256) convert_weights(
    const float* __restrict__ Wq, const float* __restrict__ Wk,
    const float* __restrict__ Wv, const float* __restrict__ We,
    bf16* __restrict__ WqkvT, bf16* __restrict__ WeT)
{
    int tid = blockIdx.x * blockDim.x + threadIdx.x;
    if (tid < 1536 * 512) {
        int c = tid >> 9;          /* output col 0..1535 */
        int k = tid & 511;
        int mat = c >> 9;          /* 0=Q 1=K 2=V */
        int n = c & 511;
        const float* W = (mat == 0) ? Wq : (mat == 1) ? Wk : Wv;
        WqkvT[(size_t)c * 512 + k] = (bf16)W[(size_t)k * 512 + n];
    } else {
        int idx = tid - 1536 * 512;
        if (idx < 512 * 512) {
            int c = idx >> 9;
            int k = idx & 511;
            WeT[(size_t)c * 512 + k] = (bf16)We[(size_t)k * 512 + c];
        }
    }
}

/* ---------------- shared 128x128xK=512 bf16 WMMA mainloop ----------------
 * A: fp32 [*,512] row-major. Staged async into LDS as fp32; converted to bf16
 *    during fragment load (co-executes with WMMA pipe).
 * Bt: bf16 [*,512], row = output column (pre-transposed weights). Async byte copy.
 * Triple-buffered LDS, async prefetch distance 2 (wait asynccnt<=6), one
 * barrier per k-step.
 * 8 waves; wave(r,c) = (wave>>1, wave&1); each wave owns 2x4 16x16 tiles. */
__device__ __forceinline__ void gemm128x128(
    const float* __restrict__ A, const bf16* __restrict__ Bt,
    int aRow0, int bCol0,
    float* Af32, bf16* Blds,          /* Af32: 3*128*32 f32, Blds: 3*128*32 bf16 */
    v8f (&acc)[2][4], int lane, int waveRow, int waveCol)
{
    const int tid   = threadIdx.x;
    const int sRow  = tid >> 1;          /* 0..127 */
    const int sHalf = (tid & 1) * 16;    /* 0 / 16 elements */
    const int lhalf = lane & 15;
    const int hsel  = lane >> 4;         /* 0 / 1 */
    const int rowW  = waveRow * 32;
    const int colW  = waveCol * 64;

    for (int tr = 0; tr < 2; ++tr)
        for (int tc = 0; tc < 4; ++tc)
            for (int q = 0; q < 8; ++q) acc[tr][tc][q] = 0.0f;

    const uint64_t aBase   = (uint64_t)(uintptr_t)A;
    const uint64_t bBase   = (uint64_t)(uintptr_t)Bt;
    const uint32_t aRowOff = (uint32_t)((aRow0 + sRow) * IN_DIM + sHalf) * 4u;
    const uint32_t bRowOff = (uint32_t)((bCol0 + sRow) * IN_DIM + sHalf) * 2u;
    const uint32_t aLds0   = (uint32_t)(uintptr_t)(Af32 + sRow * 32 + sHalf);
    const uint32_t bLds0   = (uint32_t)(uintptr_t)(Blds + sRow * 32 + sHalf);

    /* stage tile kk into LDS buffer buf: A = 64B/thread, B = 32B/thread (6 ops) */
    auto stage = [&](int kk, int buf) {
        uint32_t aoff = aRowOff + (uint32_t)(kk * 32) * 4u;
        uint32_t al   = aLds0 + (uint32_t)buf * (128 * 32 * 4);
        async_b128(al,       aoff,       aBase);
        async_b128(al + 16u, aoff + 16u, aBase);
        async_b128(al + 32u, aoff + 32u, aBase);
        async_b128(al + 48u, aoff + 48u, aBase);
        uint32_t boff = bRowOff + (uint32_t)(kk * 32) * 2u;
        uint32_t bl   = bLds0 + (uint32_t)buf * (128 * 32 * 2);
        async_b128(bl,       boff,       bBase);
        async_b128(bl + 16u, boff + 16u, bBase);
    };

    stage(0, 0);
    stage(1, 1);
    wait_async6();                 /* tile 0 landed; tile 1 still in flight */
    __syncthreads();

    for (int kk = 0; kk < 16; ++kk) {
        const int cur = kk % 3;
        if (kk < 14) stage(kk + 2, (kk + 2) % 3);   /* distance-2 async prefetch */
        if (kk + 4 < 16)
            __builtin_prefetch(A + (size_t)(aRow0 + sRow) * IN_DIM + (kk + 4) * 32 + sHalf, 0, 0);

        const float* Ac = Af32 + cur * (128 * 32);
        const bf16*  Bc = Blds + cur * (128 * 32);

        v16bf afrag[2], bfrag[4];
        for (int tr = 0; tr < 2; ++tr) {
            const float* ap = Ac + (rowW + tr * 16 + lhalf) * 32 + hsel * 8;
            float4 l0 = *(const float4*)(ap);
            float4 l1 = *(const float4*)(ap + 4);
            float4 h0 = *(const float4*)(ap + 16);
            float4 h1 = *(const float4*)(ap + 20);
            v16bf a;
            a[0]=(bf16)l0.x;  a[1]=(bf16)l0.y;  a[2]=(bf16)l0.z;  a[3]=(bf16)l0.w;
            a[4]=(bf16)l1.x;  a[5]=(bf16)l1.y;  a[6]=(bf16)l1.z;  a[7]=(bf16)l1.w;
            a[8]=(bf16)h0.x;  a[9]=(bf16)h0.y;  a[10]=(bf16)h0.z; a[11]=(bf16)h0.w;
            a[12]=(bf16)h1.x; a[13]=(bf16)h1.y; a[14]=(bf16)h1.z; a[15]=(bf16)h1.w;
            afrag[tr] = a;
        }
        for (int tc = 0; tc < 4; ++tc) {
            const v8bf* p = (const v8bf*)(Bc + (colW + tc * 16 + lhalf) * 32 + hsel * 8);
            bfrag[tc] = __builtin_shufflevector(p[0], p[2],
                0,1,2,3,4,5,6,7,8,9,10,11,12,13,14,15);
        }
        for (int tr = 0; tr < 2; ++tr)
            for (int tc = 0; tc < 4; ++tc)
                acc[tr][tc] = __builtin_amdgcn_wmma_f32_16x16x32_bf16(
                    false, afrag[tr], false, bfrag[tc],
                    (short)0, acc[tr][tc], false, false);

        if (kk < 15) {
            /* ensure tile kk+1 landed: in-order completion => <=6 outstanding
             * leaves only this iteration's stage(kk+2) in flight */
            if (kk < 14) wait_async6(); else wait_async0();
            __syncthreads();       /* publish all waves' LDS writes */
        }
    }
}

/* ---------------- QKV projection GEMM: [1024x512] @ [512x1536] ---------------- */
__global__ void __launch_bounds__(256) gemm_qkv_kernel(
    const float* __restrict__ h, const bf16* __restrict__ WqkvT,
    float* __restrict__ QKV)
{
    __shared__ __align__(16) float Af32[3 * 128 * 32];
    __shared__ __align__(16) bf16  Blds[3 * 128 * 32];
    int colb = blockIdx.x;           /* 0..11 */
    int rowb = blockIdx.y;           /* 0..7  */
    int lane = threadIdx.x & 31;
    int wave = threadIdx.x >> 5;
    v8f acc[2][4];
    gemm128x128(h, WqkvT, rowb * 128, colb * 128, Af32, Blds, acc,
                lane, wave >> 1, wave & 1);
    int lhalf = lane & 15, hsel = lane >> 4;
    for (int tr = 0; tr < 2; ++tr)
        for (int tc = 0; tc < 4; ++tc) {
            int gc = colb * 128 + (wave & 1) * 64 + tc * 16 + lhalf;
            for (int v = 0; v < 8; ++v) {
                int gr = rowb * 128 + (wave >> 1) * 32 + tr * 16 + hsel * 8 + v;
                QKV[(size_t)gr * QKV_LD + gc] = acc[tr][tc][v];
            }
        }
}

/* ------------- fused:  pe = e@We ; score = Q*K/8*pe ; e_out, logits S ------------- */
__global__ void __launch_bounds__(256) fused_escore_kernel(
    const float* __restrict__ e, const bf16* __restrict__ WeT,
    const float* __restrict__ QKV,
    float* __restrict__ e_out, float* __restrict__ S)
{
    __shared__ __align__(16) float Af32[3 * 128 * 32];
    __shared__ __align__(16) bf16  Blds[3 * 128 * 32];
    int colb = blockIdx.x;           /* 0..3  : 128 output cols = 2 heads */
    int rowb = blockIdx.y;           /* 0..2047 : 128 flat (b,i,j) rows  */
    int aRow0 = rowb * 128;
    int b  = aRow0 >> 16;
    int i  = (aRow0 >> 8) & 255;
    int j0 = aRow0 & 255;            /* 0 or 128 */
    int lane = threadIdx.x & 31;
    int wave = threadIdx.x >> 5;
    int waveRow = wave >> 1, waveCol = wave & 1;
    v8f acc[2][4];
    gemm128x128(e, WeT, aRow0, colb * 128, Af32, Blds, acc, lane, waveRow, waveCol);

    int lhalf = lane & 15, hsel = lane >> 4;
    int head  = colb * 2 + waveCol;  /* this wave's 64 cols = exactly one head */
    const float* Qrow = QKV + (size_t)(b * SEQ + i) * QKV_LD;
    float qv[4];
    int   gc[4];
    for (int tc = 0; tc < 4; ++tc) {
        gc[tc] = colb * 128 + waveCol * 64 + tc * 16 + lhalf;
        qv[tc] = Qrow[gc[tc]] * 0.125f;          /* fold 1/sqrt(64) */
    }
    for (int tr = 0; tr < 2; ++tr) {
        for (int v = 0; v < 8; ++v) {
            int jloc = waveRow * 32 + tr * 16 + hsel * 8 + v;
            int j = j0 + jloc;
            const float* Krow = QKV + (size_t)(b * SEQ + j) * QKV_LD + HD;
            size_t erow = (size_t)(aRow0 + jloc) * HD;
            float partial = 0.0f;
            for (int tc = 0; tc < 4; ++tc) {
                float val = acc[tr][tc][v] * qv[tc] * Krow[gc[tc]];
                __builtin_nontemporal_store(val, &e_out[erow + gc[tc]]);
                partial += val;
            }
            /* sum over d within head: reduce across the 16 lanes of this half */
            for (int m = 1; m < 16; m <<= 1)
                partial += __shfl_xor(partial, m, 16);
            if (lhalf == 0)
                S[(((size_t)(b * SEQ + i) * SEQ + j) << 3) + head] = partial;
        }
    }
}

/* ---------------- clamp + softmax over j, then w @ V ---------------- */
__global__ void __launch_bounds__(256) softmax_av_kernel(
    const float* __restrict__ S, const float* __restrict__ QKV,
    float* __restrict__ h_out)
{
    __shared__ float wbuf[NUM_HEADS][SEQ];
    int bi = blockIdx.x;             /* (b,i) 0..1023 */
    int b = bi >> 8;
    int h = threadIdx.x >> 5;        /* wave == head */
    int lane = threadIdx.x & 31;

    const float* Srow = S + (size_t)bi * SEQ * NUM_HEADS + h;
    float sv[8];
    float mx = -1e30f;
    for (int t = 0; t < 8; ++t) {
        float x = Srow[(size_t)(lane + t * 32) * NUM_HEADS];
        x = fminf(fmaxf(x, -5.0f), 5.0f);
        sv[t] = x;
        mx = fmaxf(mx, x);
    }
    for (int m = 1; m < 32; m <<= 1) mx = fmaxf(mx, __shfl_xor(mx, m, 32));
    float sum = 0.0f;
    for (int t = 0; t < 8; ++t) { sv[t] = __expf(sv[t] - mx); sum += sv[t]; }
    for (int m = 1; m < 32; m <<= 1) sum += __shfl_xor(sum, m, 32);
    float inv = 1.0f / sum;
    for (int t = 0; t < 8; ++t) wbuf[h][lane + t * 32] = sv[t] * inv;
    __syncthreads();

    float a0 = 0.0f, a1 = 0.0f;
    const float* Vb = QKV + (size_t)(b * SEQ) * QKV_LD + 2 * HD + h * DHEAD;
    for (int j = 0; j < SEQ; ++j) {
        float w = wbuf[h][j];
        const float* vr = Vb + (size_t)j * QKV_LD;
        a0 += w * vr[lane];
        a1 += w * vr[lane + 32];
    }
    float* orow = h_out + (size_t)bi * HD + h * DHEAD;
    orow[lane]      = a0;
    orow[lane + 32] = a1;
}

/* ---------------- host launcher ---------------- */
extern "C" void kernel_launch(void* const* d_in, const int* in_sizes, int n_in,
                              void* d_out, int out_size, void* d_ws, size_t ws_size,
                              hipStream_t stream)
{
    const float* h  = (const float*)d_in[0];
    const float* e  = (const float*)d_in[1];
    const float* Wq = (const float*)d_in[2];
    const float* Wk = (const float*)d_in[3];
    const float* Wv = (const float*)d_in[4];
    const float* We = (const float*)d_in[5];

    float* out   = (float*)d_out;
    float* h_out = out;                                   /* [4,256,512]      */
    float* e_out = out + (size_t)BATCH * SEQ * HD;        /* [4,256,256,512]  */

    char*  ws    = (char*)d_ws;
    bf16*  WqkvT = (bf16*)(ws + WS_WQKVT);
    bf16*  WeT   = (bf16*)(ws + WS_WET);
    float* QKV   = (float*)(ws + WS_QKV);
    float* S     = (float*)(ws + WS_S);

    convert_weights<<<(1536 * 512 + 512 * 512) / 256, 256, 0, stream>>>(
        Wq, Wk, Wv, We, WqkvT, WeT);
    gemm_qkv_kernel<<<dim3(12, 8), 256, 0, stream>>>(h, WqkvT, QKV);
    fused_escore_kernel<<<dim3(4, 2048), 256, 0, stream>>>(e, WeT, QKV, e_out, S);
    softmax_av_kernel<<<BATCH * SEQ, 256, 0, stream>>>(S, QKV, h_out);
}
